// GATPlayerTracker_61314953118333
// MI455X (gfx1250) — compile-verified, matrix operand order checked by reading, not run on hardware
//
#include <hip/hip_runtime.h>

#define N_NODES 20000
#define E_EDGES 320000
#define ET      (E_EDGES + N_NODES)   // with self-loops
#define F_INPUT 128
#define HID     128
#define HEADS   4
#define HC      (HEADS * HID)         // 512
#define B_GR    64
#define OUT_DIM 2
#define SLOPE   0.2f

typedef __attribute__((ext_vector_type(16))) _Float16 v16h;
typedef __attribute__((ext_vector_type(8)))  _Float16 v8h;
typedef __attribute__((ext_vector_type(4)))  _Float16 v4h;
typedef __attribute__((ext_vector_type(8)))  float    v8f;

// ---------------------------------------------------------------- utilities
__global__ void zero_f32(float* __restrict__ p, int n) {
    int i = blockIdx.x * blockDim.x + threadIdx.x;
    if (i < n) p[i] = 0.0f;
}

__global__ void to_f16(const float* __restrict__ x, _Float16* __restrict__ y, int n) {
    int i = blockIdx.x * blockDim.x + threadIdx.x;
    if (i < n) y[i] = (_Float16)x[i];
}

// WT[n*K+k] = (f16) W[k*N+n]   (row-major KxN -> row-major NxK)
__global__ void transpose_to_f16(const float* __restrict__ W, _Float16* __restrict__ WT,
                                 int K, int N) {
    int i = blockIdx.x * blockDim.x + threadIdx.x;
    if (i >= K * N) return;
    int n = i / K, k = i - n * K;
    WT[i] = (_Float16)W[(size_t)k * N + n];
}

// ------------------------------------------------------- WMMA GEMM + bias
__device__ __forceinline__ v16h load_frag(const _Float16* __restrict__ p) {
    // ISA 16-bit fragment: elements 0..7 = K +0..7, 8..15 = K +16..23 (per half-wave)
    v8h lo = *(const v8h*)(p);
    v8h hi = *(const v8h*)(p + 16);
    v16h f;
#pragma unroll
    for (int i = 0; i < 8; ++i) { f[i] = lo[i]; f[i + 8] = hi[i]; }
    return f;
}

// C[M x N] = A[M x K] * B[K x N] + bias[N]; BT = B transposed (N x K, f16).
// Block = 128 threads = 4 waves arranged 2(M) x 2(N); wave tile 16M x 64N
// (4 accumulators, A fragment reused by 4 WMMAs). Block tile 32M x 128N.
// M % 32 == 0, N % 128 == 0, K % 32 == 0. No divergence -> EXEC all ones.
__global__ void gemm_wmma_bias(const _Float16* __restrict__ A,
                               const _Float16* __restrict__ BT,
                               const float*    __restrict__ bias,
                               float*          __restrict__ C,
                               int M, int N, int K) {
    const int lane   = threadIdx.x & 31;
    const int wave   = threadIdx.x >> 5;
    const int wm     = wave >> 1, wn = wave & 1;
    const int l15    = lane & 15;
    const int half16 = lane >> 4;                 // 0 or 1
    const int mBase  = blockIdx.y * 32 + wm * 16;
    const int nBase  = blockIdx.x * 128 + wn * 64;
    const int kb0    = half16 << 3;               // K sub-offset: 0 or 8

    const _Float16* __restrict__ arow = A  + (size_t)(mBase + l15) * K + kb0;
    const _Float16* __restrict__ brow = BT + (size_t)(nBase + l15) * K + kb0;
    const size_t tstride = (size_t)16 * K;        // 16 columns of BT

    v8f acc0 = {}, acc1 = {}, acc2 = {}, acc3 = {};
    for (int k = 0; k < K; k += 32) {
        v16h a  = load_frag(arow + k);
        v16h b0 = load_frag(brow + k);
        v16h b1 = load_frag(brow + tstride + k);
        v16h b2 = load_frag(brow + 2 * tstride + k);
        v16h b3 = load_frag(brow + 3 * tstride + k);
        acc0 = __builtin_amdgcn_wmma_f32_16x16x32_f16(false, a, false, b0, (short)0, acc0, false, false);
        acc1 = __builtin_amdgcn_wmma_f32_16x16x32_f16(false, a, false, b1, (short)0, acc1, false, false);
        acc2 = __builtin_amdgcn_wmma_f32_16x16x32_f16(false, a, false, b2, (short)0, acc2, false, false);
        acc3 = __builtin_amdgcn_wmma_f32_16x16x32_f16(false, a, false, b3, (short)0, acc3, false, false);
    }

    const int mrow = half16 << 3;                 // C/D layout: VGPR v -> M = v + 8*half16
    v8f accs[4] = {acc0, acc1, acc2, acc3};
#pragma unroll
    for (int t = 0; t < 4; ++t) {
        int col  = nBase + t * 16 + l15;
        float bv = bias[col];
#pragma unroll
        for (int v = 0; v < 8; ++v)
            C[(size_t)(mBase + mrow + v) * N + col] = accs[t][v] + bv;
    }
}

// ---------------------------------------------------------------- CSR build
__device__ __forceinline__ int edge_dst_of(int e, const int* __restrict__ dst) {
    return (e < E_EDGES) ? dst[e] : (e - E_EDGES);
}
__device__ __forceinline__ int edge_src_of(int e, const int* __restrict__ src) {
    return (e < E_EDGES) ? src[e] : (e - E_EDGES);
}

__global__ void csr_count(const int* __restrict__ dst, int* __restrict__ deg) {
    int e = blockIdx.x * blockDim.x + threadIdx.x;
    if (e < ET) atomicAdd(&deg[edge_dst_of(e, dst)], 1);
}

// single-block exclusive scan (n up to ~millions; here n = 20000)
__global__ void exclusive_scan(const int* __restrict__ in, int* __restrict__ out, int n) {
    __shared__ int buf[1024];
    __shared__ int carry;
    if (threadIdx.x == 0) carry = 0;
    __syncthreads();
    for (int base = 0; base < n; base += 1024) {
        int i = base + (int)threadIdx.x;
        int v = (i < n) ? in[i] : 0;
        buf[threadIdx.x] = v;
        __syncthreads();
        for (int off = 1; off < 1024; off <<= 1) {
            int t = (threadIdx.x >= (unsigned)off) ? buf[threadIdx.x - off] : 0;
            __syncthreads();
            buf[threadIdx.x] += t;
            __syncthreads();
        }
        if (i < n) out[i] = carry + buf[threadIdx.x] - v;   // exclusive
        int total = buf[1023];
        __syncthreads();
        if (threadIdx.x == 0) carry += total;
        __syncthreads();
    }
    if (threadIdx.x == 0) out[n] = carry;
}

__global__ void csr_fill(const int* __restrict__ dst, const int* __restrict__ rowptr,
                         int* __restrict__ fillpos, int* __restrict__ eidx) {
    int e = blockIdx.x * blockDim.x + threadIdx.x;
    if (e >= ET) return;
    int d = edge_dst_of(e, dst);
    int p = atomicAdd(&fillpos[d], 1);
    eidx[rowptr[d] + p] = e;
}

// ------------------------------------------------------------ GATv2 edges
// logits[e,h] = sum_c att[h,c] * leaky_relu(xl[s,h,c] + xr[d,h,c])
__global__ void edge_logits(const float* __restrict__ xl, const float* __restrict__ xr,
                            const float* __restrict__ att,
                            const int* __restrict__ src, const int* __restrict__ dst,
                            float* __restrict__ logits, int H, int C) {
    int idx = blockIdx.x * blockDim.x + threadIdx.x;
    if (idx >= ET * H) return;
    int e = idx / H, h = idx - e * H;
    int s = edge_src_of(e, src), d = edge_dst_of(e, dst);
    const float* pl = xl + (size_t)s * H * C + h * C;
    const float* pr = xr + (size_t)d * H * C + h * C;
    const float* pa = att + h * C;
    float acc = 0.0f;
    for (int c = 0; c < C; c += 4) {
        float4 a = *(const float4*)(pl + c);
        float4 b = *(const float4*)(pr + c);
        float4 w = *(const float4*)(pa + c);
        float v;
        v = a.x + b.x; v = v > 0.0f ? v : SLOPE * v; acc += w.x * v;
        v = a.y + b.y; v = v > 0.0f ? v : SLOPE * v; acc += w.y * v;
        v = a.z + b.z; v = v > 0.0f ? v : SLOPE * v; acc += w.z * v;
        v = a.w + b.w; v = v > 0.0f ? v : SLOPE * v; acc += w.w * v;
    }
    logits[idx] = acc;
}

// per-(dst,head) segment softmax over in-edges; normalized alpha written in place.
// Conflict-free: each edge belongs to exactly one dst.
__global__ void softmax_gather(float* __restrict__ logits, const int* __restrict__ rowptr,
                               const int* __restrict__ eidx, int H) {
    int idx = blockIdx.x * blockDim.x + threadIdx.x;
    if (idx >= N_NODES * H) return;
    int d = idx / H, h = idx - d * H;
    int b = rowptr[d], en = rowptr[d + 1];      // en > b (self-loop guaranteed)
    float mx = -3.4e38f;
    for (int i = b; i < en; ++i) mx = fmaxf(mx, logits[eidx[i] * H + h]);
    float sum = 0.0f;
    for (int i = b; i < en; ++i) {
        float a = expf(logits[eidx[i] * H + h] - mx);
        logits[eidx[i] * H + h] = a;
        sum += a;
    }
    float inv = 1.0f / sum;
    for (int i = b; i < en; ++i) logits[eidx[i] * H + h] *= inv;
}

__device__ __forceinline__ float elu(float v) { return v > 0.0f ? v : (expf(v) - 1.0f); }

// per-(dst, 4 columns) gather of alpha * xl[src], fused bias + ELU.
// mode 0: write f16 (next layer GEMM input). mode 1: write f32 (pooling input).
__global__ void gather_finalize(const float* __restrict__ xl, const float* __restrict__ alpha,
                                const int* __restrict__ src, const int* __restrict__ rowptr,
                                const int* __restrict__ eidx, const float* __restrict__ bias,
                                _Float16* __restrict__ out16, float* __restrict__ out32,
                                int H, int C, int mode) {
    int HCn = H * C;
    int idx = blockIdx.x * blockDim.x + threadIdx.x;
    if (idx >= N_NODES * (HCn >> 2)) return;
    int d  = idx / (HCn >> 2);
    int j  = (idx - d * (HCn >> 2)) << 2;        // column group (4 floats, one head)
    int h  = j / C;
    float4 acc = make_float4(0.f, 0.f, 0.f, 0.f);
    int b = rowptr[d], en = rowptr[d + 1];
    for (int i = b; i < en; ++i) {
        int e = eidx[i];
        int s = edge_src_of(e, src);
        float a  = alpha[e * H + h];
        float4 v = *(const float4*)(xl + (size_t)s * HCn + j);
        acc.x += a * v.x; acc.y += a * v.y; acc.z += a * v.z; acc.w += a * v.w;
    }
    float o0 = elu(acc.x + bias[j]);
    float o1 = elu(acc.y + bias[j + 1]);
    float o2 = elu(acc.z + bias[j + 2]);
    float o3 = elu(acc.w + bias[j + 3]);
    size_t o = (size_t)d * HCn + j;
    if (mode == 0) {
        v4h p; p[0] = (_Float16)o0; p[1] = (_Float16)o1; p[2] = (_Float16)o2; p[3] = (_Float16)o3;
        *(v4h*)(out16 + o) = p;
    } else {
        *(float4*)(out32 + o) = make_float4(o0, o1, o2, o3);
    }
}

// --------------------------------------------------------------- pooling/MLP
__global__ void pool_accum(const float* __restrict__ h3, const int* __restrict__ batch,
                           float* __restrict__ pooled, float* __restrict__ cnt) {
    int idx = blockIdx.x * blockDim.x + threadIdx.x;
    if (idx >= N_NODES * HID) return;
    int n = idx >> 7, j = idx & 127;
    int b = batch[n];
    atomicAdd(&pooled[(size_t)b * HID + j], h3[idx]);
    if (j == 0) atomicAdd(&cnt[b], 1.0f);
}

__global__ void mlp1(const float* __restrict__ pooled, const float* __restrict__ cnt,
                     const float* __restrict__ W, const float* __restrict__ b,
                     float* __restrict__ hidden) {
    int idx = blockIdx.x * blockDim.x + threadIdx.x;
    if (idx >= B_GR * HID) return;
    int g = idx >> 7, k = idx & 127;
    float invc = 1.0f / fmaxf(cnt[g], 1.0f);
    float s = b[k];
    for (int j = 0; j < HID; ++j)
        s += pooled[(size_t)g * HID + j] * invc * W[(size_t)j * HID + k];
    hidden[idx] = fmaxf(s, 0.0f);
}

__global__ void mlp2(const float* __restrict__ hidden, const float* __restrict__ W,
                     const float* __restrict__ b, float* __restrict__ out) {
    int idx = blockIdx.x * blockDim.x + threadIdx.x;
    if (idx >= B_GR * OUT_DIM) return;
    int g = idx / OUT_DIM, o = idx - g * OUT_DIM;
    float s = b[o];
    for (int k = 0; k < HID; ++k)
        s += hidden[(size_t)g * HID + k] * W[(size_t)k * OUT_DIM + o];
    out[idx] = s;
}

// ------------------------------------------------------------------- launch
static inline size_t al256(size_t x) { return (x + 255) & ~(size_t)255; }

extern "C" void kernel_launch(void* const* d_in, const int* in_sizes, int n_in,
                              void* d_out, int out_size, void* d_ws, size_t ws_size,
                              hipStream_t stream) {
    const float* x     = (const float*)d_in[0];
    const int*   esrc  = (const int*)d_in[1];
    const int*   edst  = (const int*)d_in[2];
    const int*   batch = (const int*)d_in[3];
    const float* Wl[3]   = {(const float*)d_in[4],  (const float*)d_in[10], (const float*)d_in[16]};
    const float* Wr[3]   = {(const float*)d_in[5],  (const float*)d_in[11], (const float*)d_in[17]};
    const float* bl[3]   = {(const float*)d_in[6],  (const float*)d_in[12], (const float*)d_in[18]};
    const float* br[3]   = {(const float*)d_in[7],  (const float*)d_in[13], (const float*)d_in[19]};
    const float* att[3]  = {(const float*)d_in[8],  (const float*)d_in[14], (const float*)d_in[20]};
    const float* bias[3] = {(const float*)d_in[9],  (const float*)d_in[15], (const float*)d_in[21]};
    const float* Wm1 = (const float*)d_in[22];
    const float* bm1 = (const float*)d_in[23];
    const float* Wm2 = (const float*)d_in[24];
    const float* bm2 = (const float*)d_in[25];
    float* out = (float*)d_out;

    // ----- workspace layout (~110 MB)
    char* ws = (char*)d_ws;
    size_t off = 0;
    _Float16* A16 = (_Float16*)(ws + off); off = al256(off + (size_t)N_NODES * HC * 2);
    _Float16* WTl = (_Float16*)(ws + off); off = al256(off + (size_t)HC * HC * 2);
    _Float16* WTr = (_Float16*)(ws + off); off = al256(off + (size_t)HC * HC * 2);
    float* xl     = (float*)(ws + off);    off = al256(off + (size_t)N_NODES * HC * 4);
    float* xr     = (float*)(ws + off);    off = al256(off + (size_t)N_NODES * HC * 4);
    float* logits = (float*)(ws + off);    off = al256(off + (size_t)ET * HEADS * 4);
    int* deg      = (int*)(ws + off);      off = al256(off + (size_t)N_NODES * 4);
    int* fillpos  = (int*)(ws + off);      off = al256(off + (size_t)N_NODES * 4);
    int* rowptr   = (int*)(ws + off);      off = al256(off + (size_t)(N_NODES + 1) * 4);
    int* eidx     = (int*)(ws + off);      off = al256(off + (size_t)ET * 4);
    float* pooled = (float*)(ws + off);    off = al256(off + (size_t)B_GR * HID * 4);
    float* cnt    = (float*)(ws + off);    off = al256(off + (size_t)B_GR * 4);
    float* hidden = (float*)(ws + off);    off = al256(off + (size_t)B_GR * HID * 4);
    float* h3     = xr;   // xr is dead after layer-3 softmax; reuse for f32 output

    const int T = 256;
    #define GRID(n) dim3(((n) + T - 1) / T)

    // ----- CSR by destination (graph is identical for all 3 layers)
    zero_f32<<<GRID(N_NODES), T, 0, stream>>>((float*)deg, N_NODES);
    zero_f32<<<GRID(N_NODES), T, 0, stream>>>((float*)fillpos, N_NODES);
    csr_count<<<GRID(ET), T, 0, stream>>>(edst, deg);
    exclusive_scan<<<1, 1024, 0, stream>>>(deg, rowptr, N_NODES);
    csr_fill<<<GRID(ET), T, 0, stream>>>(edst, rowptr, fillpos, eidx);

    const int Kdim[3] = {F_INPUT, HC, HC};
    const int Hh[3]   = {HEADS, HEADS, 1};
    const int Nout[3] = {HC, HC, HID};     // H*C per layer

    // layer-1 input -> f16
    to_f16<<<GRID(N_NODES * F_INPUT), T, 0, stream>>>(x, A16, N_NODES * F_INPUT);

    for (int L = 0; L < 3; ++L) {
        const int K = Kdim[L], H = Hh[L], NO = Nout[L], C = HID;

        // weights -> f16, transposed (N x K)
        transpose_to_f16<<<GRID(K * NO), T, 0, stream>>>(Wl[L], WTl, K, NO);
        transpose_to_f16<<<GRID(K * NO), T, 0, stream>>>(Wr[L], WTr, K, NO);

        // xl = A @ Wl + bl ; xr = A @ Wr + br   (WMMA)
        dim3 ggrid(NO / 128, N_NODES / 32);
        gemm_wmma_bias<<<ggrid, 128, 0, stream>>>(A16, WTl, bl[L], xl, N_NODES, NO, K);
        gemm_wmma_bias<<<ggrid, 128, 0, stream>>>(A16, WTr, br[L], xr, N_NODES, NO, K);

        // attention: per-edge logits, then gather-style softmax + aggregation
        edge_logits<<<GRID(ET * H), T, 0, stream>>>(xl, xr, att[L], esrc, edst, logits, H, C);
        softmax_gather<<<GRID(N_NODES * H), T, 0, stream>>>(logits, rowptr, eidx, H);
        gather_finalize<<<GRID(N_NODES * (NO / 4)), T, 0, stream>>>(
            xl, logits, esrc, rowptr, eidx, bias[L],
            A16, h3, H, C, (L < 2) ? 0 : 1);
    }

    // global mean pool + MLP head
    zero_f32<<<GRID(B_GR * HID), T, 0, stream>>>(pooled, B_GR * HID);
    zero_f32<<<GRID(B_GR), T, 0, stream>>>(cnt, B_GR);
    pool_accum<<<GRID(N_NODES * HID), T, 0, stream>>>(h3, batch, pooled, cnt);
    mlp1<<<GRID(B_GR * HID), T, 0, stream>>>(pooled, cnt, Wm1, bm1, hidden);
    mlp2<<<GRID(B_GR * OUT_DIM), T, 0, stream>>>(hidden, Wm2, bm2, out);

    #undef GRID
}
